// EGNNDynamics_80118319940178
// MI455X (gfx1250) — matrix operand order repaired and checked by reading, not compiled, not run on hardware
//
#include <hip/hip_runtime.h>
#include <hip/hip_bf16.h>

typedef _Float16 v16h __attribute__((ext_vector_type(16)));
typedef _Float16 v8h  __attribute__((ext_vector_type(8)));
typedef float    v8f  __attribute__((ext_vector_type(8)));

#define HID    256
#define NNODES 10000
#define NEDGES 320000
#define LSTR   264   // padded LDS row stride in halfs: 16B-aligned, conflict-free

// ---------------------------------------------------------------- helpers

__device__ __forceinline__ float fast_silu(float v) {
  return v / (1.0f + __expf(-v));
}
__device__ __forceinline__ float fast_sigmoid(float v) {
  return 1.0f / (1.0f + __expf(-v));
}
__device__ __forceinline__ v16h cat8(v8h a, v8h b) {
  return __builtin_shufflevector(a, b, 0,1,2,3,4,5,6,7,8,9,10,11,12,13,14,15);
}

// Core: acc[MT*16] (MT x 16x256 C tiles, f32) = stage2( silu(stage1(A)) )
//   stage1: K=512 from {a0base[t] (K 0..255), a1base[t] (K 256..511)} @ w1p
//           + optional rank-2 edge_attr tail (w1tail rows 512/513, f32) + b1 + SiLU
//   transpose via padded LDS (f16), stage2: K=256 @ w2p + b2 (+SiLU if silu2)
// Each B fragment load feeds MT WMMAs; B is explicitly double-buffered so the
// next fragment's loads are in flight while the current pair of WMMAs issues.
template <int MT>
__device__ __forceinline__ void egnn_mlp_core(
    const _Float16* const* a0base,     // [MT] per-lane row base, K 0..255
    const _Float16* const* a1base,     // [MT] per-lane row base, K 256..511
    const v16h*  __restrict__ w1p,
    const float* __restrict__ w1tail,  // rows 512,513 of w1 (f32) or nullptr
    const float* __restrict__ b1,
    const v16h*  __restrict__ w2p,
    const float* __restrict__ b2,
    const float* __restrict__ ea,      // edge_attr tile base [MT*16][2] or nullptr
    _Float16* __restrict__ lds,        // this wave's (MT*16) x LSTR f16 region
    int lane, v8f* __restrict__ acc, bool silu2)
{
  const int m   = lane & 15;
  const int grp = lane >> 4;

  v8f zero = {0.f,0.f,0.f,0.f,0.f,0.f,0.f,0.f};
#pragma unroll
  for (int t = 0; t < MT * 16; ++t) acc[t] = zero;

  // ---- stage 1: 16 K-tiles of 32 --------------------------------------
#pragma unroll
  for (int kt = 0; kt < 16; ++kt) {
    v16h a[MT];
#pragma unroll
    for (int t = 0; t < MT; ++t) {
      const _Float16* ap =
          ((kt < 8) ? a0base[t] : a1base[t]) + (kt & 7) * 32 + grp * 8;
      a[t] = cat8(*(const v8h*)(ap), *(const v8h*)(ap + 16));
    }
    const v16h* bp = w1p + (size_t)(kt * 16) * 32 + lane;
    v16h b = bp[0];
#pragma unroll
    for (int nt = 0; nt < 16; ++nt) {
      v16h bn = b;
      if (nt < 15) bn = bp[(nt + 1) * 32];     // prefetch next B fragment
#pragma unroll
      for (int t = 0; t < MT; ++t) {
        acc[t * 16 + nt] = __builtin_amdgcn_wmma_f32_16x16x32_f16(
            false, a[t], false, b, (short)0, acc[t * 16 + nt], false, false);
      }
      b = bn;
    }
  }

  // ---- rank-2 edge_attr tail + bias + SiLU -> LDS (f16) ----------------
  float eaA[MT][8], eaB[MT][8];
  if (ea) {
#pragma unroll
    for (int t = 0; t < MT; ++t)
#pragma unroll
      for (int g = 0; g < 8; ++g) {
        eaA[t][g] = ea[(t * 16 + grp * 8 + g) * 2 + 0];
        eaB[t][g] = ea[(t * 16 + grp * 8 + g) * 2 + 1];
      }
  }
#pragma unroll
  for (int nt = 0; nt < 16; ++nt) {
    const int col = nt * 16 + m;
    const float bias = b1[col];
    float t0 = 0.f, t1 = 0.f;
    if (ea) { t0 = w1tail[col]; t1 = w1tail[HID + col]; }
#pragma unroll
    for (int t = 0; t < MT; ++t) {
#pragma unroll
      for (int g = 0; g < 8; ++g) {
        float v = acc[t * 16 + nt][g] + bias;
        if (ea) v += eaA[t][g] * t0 + eaB[t][g] * t1;
        v = fast_silu(v);
        lds[(t * 16 + grp * 8 + g) * LSTR + col] = (_Float16)v;
      }
      acc[t * 16 + nt] = zero;
    }
  }
  __syncthreads();

  // ---- stage 2: 8 K-tiles of 32, A from LDS ----------------------------
#pragma unroll
  for (int kt = 0; kt < 8; ++kt) {
    v16h a[MT];
#pragma unroll
    for (int t = 0; t < MT; ++t) {
      const _Float16* ap = lds + (t * 16 + m) * LSTR + kt * 32 + grp * 8;
      a[t] = cat8(*(const v8h*)(ap), *(const v8h*)(ap + 16));
    }
    const v16h* bp = w2p + (size_t)(kt * 16) * 32 + lane;
    v16h b = bp[0];
#pragma unroll
    for (int nt = 0; nt < 16; ++nt) {
      v16h bn = b;
      if (nt < 15) bn = bp[(nt + 1) * 32];
#pragma unroll
      for (int t = 0; t < MT; ++t) {
        acc[t * 16 + nt] = __builtin_amdgcn_wmma_f32_16x16x32_f16(
            false, a[t], false, b, (short)0, acc[t * 16 + nt], false, false);
      }
      b = bn;
    }
  }
#pragma unroll
  for (int nt = 0; nt < 16; ++nt) {
    const int col = nt * 16 + m;
    const float bias = b2[col];
#pragma unroll
    for (int t = 0; t < MT; ++t)
#pragma unroll
      for (int g = 0; g < 8; ++g) {
        float v = acc[t * 16 + nt][g] + bias;
        acc[t * 16 + nt][g] = silu2 ? fast_silu(v) : v;
      }
  }
}

// ---------------------------------------------------------------- prep

// Pack f32 [K x 256] row-major weight into 16-bit B-fragment layout:
// out[((kt*16+nt)*32+lane)*16 + j] = W[kt*32 + (lane>>4)*8 + j + (j&8)][nt*16 + (lane&15)]
__global__ void egnn_pack_b(const float* __restrict__ W, int K,
                            _Float16* __restrict__ out) {
  int t = blockIdx.x * blockDim.x + threadIdx.x;
  int total = (K >> 5) * 16 * 32;
  if (t >= total) return;
  int lane = t & 31;
  int nt   = (t >> 5) & 15;
  int kt   = t >> 9;
  int col  = nt * 16 + (lane & 15);
  int kb   = kt * 32 + (lane >> 4) * 8;
  _Float16* o = out + (size_t)t * 16;
#pragma unroll
  for (int j = 0; j < 16; ++j)
    o[j] = (_Float16)W[(size_t)(kb + j + (j & 8)) * HID + col];
}

__global__ void egnn_f32_to_f16(const float* __restrict__ src,
                                _Float16* __restrict__ dst, int n, float scale) {
  int i = blockIdx.x * blockDim.x + threadIdx.x;
  if (i < n) dst[i] = (_Float16)(src[i] * scale);
}

__global__ void egnn_zero_f32(float* __restrict__ p, int n) {
  int i = blockIdx.x * blockDim.x + threadIdx.x;
  if (i < n) p[i] = 0.0f;
}

// ---------------------------------------------------------------- GCL edge
// 2 waves/block, 32 edges/wave (two M-tiles) -> each B fragment feeds 2 WMMAs.

__global__ __launch_bounds__(64, 1) void egnn_edge_kernel(
    const _Float16* __restrict__ h16,
    const int* __restrict__ rows, const int* __restrict__ cols,
    const float* __restrict__ edge_attr,
    const v16h* __restrict__ w1p, const float* __restrict__ w1tail,
    const float* __restrict__ b1,
    const v16h* __restrict__ w2p, const float* __restrict__ b2,
    const float* __restrict__ aw, const float* __restrict__ ab,
    float* __restrict__ agg)
{
  __shared__ _Float16 sh[2 * 32 * LSTR];
  const int wave = threadIdx.x >> 5;
  const int lane = threadIdx.x & 31;
  const int m    = lane & 15;
  const int grp  = lane >> 4;
  const int e0   = (blockIdx.x * 2 + wave) * 32;

  const _Float16* a0[2];
  const _Float16* a1[2];
#pragma unroll
  for (int t = 0; t < 2; ++t) {
    a0[t] = h16 + (size_t)rows[e0 + t * 16 + m] * HID;
    a1[t] = h16 + (size_t)cols[e0 + t * 16 + m] * HID;
  }

  v8f acc[32];
  egnn_mlp_core<2>(a0, a1, w1p, w1tail, b1, w2p, b2,
                   edge_attr + (size_t)e0 * 2,
                   sh + wave * 32 * LSTR, lane, acc, /*silu2=*/true);

  const float abv = ab[0];
#pragma unroll
  for (int t = 0; t < 2; ++t) {
    // attention: att[row] = sigmoid(sum_col mij*aw + ab); cross-lane reduce
    float p[8];
#pragma unroll
    for (int g = 0; g < 8; ++g) p[g] = 0.f;
#pragma unroll
    for (int nt = 0; nt < 16; ++nt) {
      float w = aw[nt * 16 + m];
#pragma unroll
      for (int g = 0; g < 8; ++g) p[g] += acc[t * 16 + nt][g] * w;
    }
#pragma unroll
    for (int off = 1; off < 16; off <<= 1) {
#pragma unroll
      for (int g = 0; g < 8; ++g) p[g] += __shfl_xor(p[g], off, 32);
    }
    float att[8];
#pragma unroll
    for (int g = 0; g < 8; ++g) att[g] = fast_sigmoid(p[g] + abv);

    int nr[8];
#pragma unroll
    for (int g = 0; g < 8; ++g) nr[g] = rows[e0 + t * 16 + grp * 8 + g];

    // scatter ef = mij*att into agg (L2-resident f32 atomics)
#pragma unroll
    for (int nt = 0; nt < 16; ++nt) {
      const int col = nt * 16 + m;
#pragma unroll
      for (int g = 0; g < 8; ++g) {
        float ef = acc[t * 16 + nt][g] * att[g];
        (void)__hip_atomic_fetch_add(&agg[(size_t)nr[g] * HID + col], ef,
                                     __ATOMIC_RELAXED, __HIP_MEMORY_SCOPE_AGENT);
      }
    }
  }
}

// ---------------------------------------------------------------- GCL node

__global__ __launch_bounds__(32) void egnn_node_kernel(
    _Float16* __restrict__ h16,            // in: current h16; out: updated in place
    const _Float16* __restrict__ agg16,    // (agg/100) as f16
    const float* __restrict__ h_in,        // f32 residual input
    const v16h* __restrict__ w1p, const float* __restrict__ b1,
    const v16h* __restrict__ w2p, const float* __restrict__ b2,
    float* __restrict__ h_out)
{
  __shared__ _Float16 sh[16 * LSTR];
  const int lane = threadIdx.x & 31;
  const int m    = lane & 15;
  const int grp  = lane >> 4;
  const int n0   = blockIdx.x * 16;
  const int node = n0 + m;

  const _Float16* a0[1] = {h16 + (size_t)node * HID};
  const _Float16* a1[1] = {agg16 + (size_t)node * HID};

  v8f acc[16];
  egnn_mlp_core<1>(a0, a1, w1p, nullptr, b1, w2p, b2, nullptr,
                   sh, lane, acc, /*silu2=*/false);

#pragma unroll
  for (int nt = 0; nt < 16; ++nt) {
    const int col = nt * 16 + m;
#pragma unroll
    for (int g = 0; g < 8; ++g) {
      const int r = n0 + grp * 8 + g;
      float v = acc[nt][g] + h_in[(size_t)r * HID + col];
      h_out[(size_t)r * HID + col] = v;
      h16[(size_t)r * HID + col]   = (_Float16)v;
    }
  }
}

// ---------------------------------------------------------------- coord

__global__ __launch_bounds__(64, 1) void egnn_coord_kernel(
    const _Float16* __restrict__ h16,
    const int* __restrict__ rows, const int* __restrict__ cols,
    const float* __restrict__ edge_attr, const float* __restrict__ x,
    const v16h* __restrict__ w1p, const float* __restrict__ w1tail,
    const float* __restrict__ b1,
    const v16h* __restrict__ w2p, const float* __restrict__ b2,
    const float* __restrict__ w3,
    float* __restrict__ xagg)              // [N][4] padded
{
  __shared__ _Float16 sh[2 * 32 * LSTR];
  __shared__ float phis[2][32];
  const int wave = threadIdx.x >> 5;
  const int lane = threadIdx.x & 31;
  const int m    = lane & 15;
  const int grp  = lane >> 4;
  const int e0   = (blockIdx.x * 2 + wave) * 32;

  const _Float16* a0[2];
  const _Float16* a1[2];
#pragma unroll
  for (int t = 0; t < 2; ++t) {
    a0[t] = h16 + (size_t)rows[e0 + t * 16 + m] * HID;
    a1[t] = h16 + (size_t)cols[e0 + t * 16 + m] * HID;
  }

  v8f acc[32];
  egnn_mlp_core<2>(a0, a1, w1p, w1tail, b1, w2p, b2,
                   edge_attr + (size_t)e0 * 2,
                   sh + wave * 32 * LSTR, lane, acc, /*silu2=*/true);

  // phi = mij @ cw3 (scalar per edge), per M-tile
#pragma unroll
  for (int t = 0; t < 2; ++t) {
    float p[8];
#pragma unroll
    for (int g = 0; g < 8; ++g) p[g] = 0.f;
#pragma unroll
    for (int nt = 0; nt < 16; ++nt) {
      float w = w3[nt * 16 + m];
#pragma unroll
      for (int g = 0; g < 8; ++g) p[g] += acc[t * 16 + nt][g] * w;
    }
#pragma unroll
    for (int off = 1; off < 16; off <<= 1) {
#pragma unroll
      for (int g = 0; g < 8; ++g) p[g] += __shfl_xor(p[g], off, 32);
    }
    if (m == 0) {
#pragma unroll
      for (int g = 0; g < 8; ++g) phis[wave][t * 16 + grp * 8 + g] = p[g];
    }
  }
  __syncthreads();

  // all 32 lanes: one edge each
  {
    const int e    = e0 + lane;
    const float ph = phis[wave][lane];
    const int r = rows[e], c = cols[e];
    float dx = x[r * 3 + 0] - x[c * 3 + 0];
    float dy = x[r * 3 + 1] - x[c * 3 + 1];
    float dz = x[r * 3 + 2] - x[c * 3 + 2];
    float nrm = __builtin_sqrtf(dx * dx + dy * dy + dz * dz + 1e-8f);
    float s = ph / (nrm + 1.0f);
    (void)__hip_atomic_fetch_add(&xagg[(size_t)r * 4 + 0], dx * s,
                                 __ATOMIC_RELAXED, __HIP_MEMORY_SCOPE_AGENT);
    (void)__hip_atomic_fetch_add(&xagg[(size_t)r * 4 + 1], dy * s,
                                 __ATOMIC_RELAXED, __HIP_MEMORY_SCOPE_AGENT);
    (void)__hip_atomic_fetch_add(&xagg[(size_t)r * 4 + 2], dz * s,
                                 __ATOMIC_RELAXED, __HIP_MEMORY_SCOPE_AGENT);
  }
}

__global__ void egnn_x_final(const float* __restrict__ x,
                             const float* __restrict__ xagg,
                             float* __restrict__ xout, int n) {
  int i = blockIdx.x * blockDim.x + threadIdx.x;
  if (i < n) {
    int node = i / 3, d = i - node * 3;
    xout[i] = x[i] + xagg[(size_t)node * 4 + d] * 0.01f;
  }
}

// ---------------------------------------------------------------- launch

extern "C" void kernel_launch(void* const* d_in, const int* in_sizes, int n_in,
                              void* d_out, int out_size, void* d_ws, size_t ws_size,
                              hipStream_t stream) {
  (void)in_sizes; (void)n_in; (void)out_size; (void)ws_size;

  const float* h  = (const float*)d_in[0];
  const float* x  = (const float*)d_in[1];
  const int*   ei = (const int*)d_in[2];
  const float* ea = (const float*)d_in[3];
  const int* rows = ei;
  const int* cols = ei + NEDGES;

  const float* ew1[2] = {(const float*)d_in[4],  (const float*)d_in[14]};
  const float* eb1[2] = {(const float*)d_in[5],  (const float*)d_in[15]};
  const float* ew2[2] = {(const float*)d_in[6],  (const float*)d_in[16]};
  const float* eb2[2] = {(const float*)d_in[7],  (const float*)d_in[17]};
  const float* nw1[2] = {(const float*)d_in[8],  (const float*)d_in[18]};
  const float* nb1[2] = {(const float*)d_in[9],  (const float*)d_in[19]};
  const float* nw2[2] = {(const float*)d_in[10], (const float*)d_in[20]};
  const float* nb2[2] = {(const float*)d_in[11], (const float*)d_in[21]};
  const float* aw[2]  = {(const float*)d_in[12], (const float*)d_in[22]};
  const float* ab[2]  = {(const float*)d_in[13], (const float*)d_in[23]};
  const float* cw1 = (const float*)d_in[24];
  const float* cb1 = (const float*)d_in[25];
  const float* cw2 = (const float*)d_in[26];
  const float* cb2 = (const float*)d_in[27];
  const float* cw3 = (const float*)d_in[28];

  // ---- carve workspace -------------------------------------------------
  char* ws = (char*)d_ws;
  size_t off = 0;
  auto carve = [&](size_t bytes) -> void* {
    void* p = ws + off;
    off += (bytes + 255) & ~(size_t)255;
    return p;
  };
  const size_t PACK512 = (size_t)16 * 16 * 32 * 16 * 2;  // 262144 B
  const size_t PACK256 = (size_t)8  * 16 * 32 * 16 * 2;  // 131072 B

  _Float16* h16   = (_Float16*)carve((size_t)NNODES * HID * 2);
  float*    agg   = (float*)   carve((size_t)NNODES * HID * 4);
  _Float16* agg16 = (_Float16*)carve((size_t)NNODES * HID * 2);
  float*    h1    = (float*)   carve((size_t)NNODES * HID * 4);
  float*    xagg  = (float*)   carve((size_t)NNODES * 4 * 4);
  _Float16* ew1p[2], *ew2p[2], *nw1p[2], *nw2p[2];
  for (int l = 0; l < 2; ++l) {
    ew1p[l] = (_Float16*)carve(PACK512);
    ew2p[l] = (_Float16*)carve(PACK256);
    nw1p[l] = (_Float16*)carve(PACK512);
    nw2p[l] = (_Float16*)carve(PACK256);
  }
  _Float16* cw1p = (_Float16*)carve(PACK512);
  _Float16* cw2p = (_Float16*)carve(PACK256);

  float* hout = (float*)d_out;
  float* xout = (float*)d_out + (size_t)NNODES * HID;

  // ---- pack weights & convert h ---------------------------------------
  for (int l = 0; l < 2; ++l) {
    egnn_pack_b<<<32, 256, 0, stream>>>(ew1[l], 512, ew1p[l]);
    egnn_pack_b<<<16, 256, 0, stream>>>(ew2[l], 256, ew2p[l]);
    egnn_pack_b<<<32, 256, 0, stream>>>(nw1[l], 512, nw1p[l]);
    egnn_pack_b<<<16, 256, 0, stream>>>(nw2[l], 256, nw2p[l]);
  }
  egnn_pack_b<<<32, 256, 0, stream>>>(cw1, 512, cw1p);
  egnn_pack_b<<<16, 256, 0, stream>>>(cw2, 256, cw2p);

  const int HN = NNODES * HID;             // 2,560,000
  egnn_f32_to_f16<<<HN / 256, 256, 0, stream>>>(h, h16, HN, 1.0f);

  // ---- two GCL layers --------------------------------------------------
  const int EBLK = NEDGES / 64;            // 5000 blocks x 2 waves x 32 edges
  const float* hin = h;
  for (int l = 0; l < 2; ++l) {
    egnn_zero_f32<<<HN / 256, 256, 0, stream>>>(agg, HN);
    egnn_edge_kernel<<<EBLK, 64, 0, stream>>>(
        h16, rows, cols, ea,
        (const v16h*)ew1p[l], ew1[l] + (size_t)512 * HID, eb1[l],
        (const v16h*)ew2p[l], eb2[l], aw[l], ab[l], agg);
    egnn_f32_to_f16<<<HN / 256, 256, 0, stream>>>(agg, agg16, HN, 0.01f);
    egnn_node_kernel<<<NNODES / 16, 32, 0, stream>>>(
        h16, agg16, hin,
        (const v16h*)nw1p[l], nb1[l], (const v16h*)nw2p[l], nb2[l],
        (l == 0) ? h1 : hout);
    hin = h1;
  }

  // ---- equivariant coordinate update -----------------------------------
  egnn_zero_f32<<<(NNODES * 4 + 255) / 256, 256, 0, stream>>>(xagg, NNODES * 4);
  egnn_coord_kernel<<<EBLK, 64, 0, stream>>>(
      h16, rows, cols, ea, x,
      (const v16h*)cw1p, cw1 + (size_t)512 * HID, cb1,
      (const v16h*)cw2p, cb2, cw3, xagg);
  egnn_x_final<<<(NNODES * 3 + 255) / 256, 256, 0, stream>>>(
      x, xagg, xout, NNODES * 3);
}
